// CustomMask2Former_27298812133963
// MI455X (gfx1250) — matrix operand order repaired and checked by reading, not compiled, not run on hardware
//
#include <hip/hip_runtime.h>
#include <hip/hip_bf16.h>
#include <math.h>

typedef __attribute__((ext_vector_type(16))) _Float16 v16h;
typedef __attribute__((ext_vector_type(8)))  float    v8f;
typedef __attribute__((ext_vector_type(4)))  unsigned int v4u;
typedef __attribute__((ext_vector_type(8)))  int      v8i_t;
typedef __attribute__((ext_vector_type(4)))  int      v4i_t;

#define DEV __device__ __forceinline__

// ---------------- WMMA helpers (CDNA5 16x16x32 f16 -> f32) ----------------

DEV v8f wmma32(const v16h& a, const v16h& b, const v8f& c) {
  return __builtin_amdgcn_wmma_f32_16x16x32_f16(false, a, false, b, (short)0, c,
                                                false, false);
}

// A (16x32, MxK): lane l holds row M=l&15; K slots = kb..kb+7, kb+16..kb+23, kb=(l>>4)*8
template <typename F> DEV v16h frag_a(F f) {
  v16h a;
  const int l = threadIdx.x & 31;
  const int m = l & 15;
  const int kb = (l >> 4) * 8;
#pragma unroll
  for (int j = 0; j < 16; ++j) {
    const int k = kb + j + ((j < 8) ? 0 : 8);
    a[j] = (_Float16)f(m, k);
  }
  return a;
}

// B (32x16, KxN): lane l holds col N=l&15; K slots = kb..kb+7, kb+16..kb+23, kb=(l>>4)*8
template <typename F> DEV v16h frag_b(F f) {
  v16h b;
  const int l = threadIdx.x & 31;
  const int n = l & 15;
  const int kb = (l >> 4) * 8;
#pragma unroll
  for (int j = 0; j < 16; ++j) b[j] = (_Float16)f(kb + j + ((j < 8) ? 0 : 8), n);
  return b;
}

// C/D (16x16 f32): VGPR r, lane l -> M = r + 8*(l>>4), N = l&15
template <typename F> DEV void frag_c_store(const v8f& c, F f) {
  const int l = threadIdx.x & 31;
  const int n = l & 15;
  const int mb = (l >> 4) * 8;
#pragma unroll
  for (int r = 0; r < 8; ++r) f(mb + r, n, c[r]);
}

// ---------------- implicit-GEMM conv + BN + ReLU (NHWC out) ----------------

__global__ void k_conv(const float* __restrict__ in, int Hin, int Win, int Cin,
                       int in_nchw, const float* __restrict__ w,
                       const float* __restrict__ bias, const float* __restrict__ gam,
                       const float* __restrict__ bet, const float* __restrict__ mu,
                       const float* __restrict__ var, float* __restrict__ outp,
                       int Hout, int Wout, int Cout, int stride, int Bn) {
  const int M = Bn * Hout * Wout;
  const int K = 9 * Cin;
  const int m0 = blockIdx.x * 16, n0 = blockIdx.y * 16;
  const int l = threadIdx.x & 31;

  const int mrowA = m0 + (l & 15);
  const bool mokA = mrowA < M;
  const int mmA = mokA ? mrowA : 0;
  const int sA = mmA % (Hout * Wout);
  const int biA = mmA / (Hout * Wout);
  const int oh = sA / Wout, owi = sA % Wout;

  v8f acc = {};
  for (int k0 = 0; k0 < K; k0 += 32) {
    v16h a = frag_a([&](int m, int k) -> float {
      (void)m;
      int kk = k0 + k;
      if (!mokA || kk >= K) return 0.f;
      int pidx = kk / Cin, cin = kk - pidx * Cin;
      int kh = pidx / 3, kw = pidx - kh * 3;
      int ih = oh * stride + kh - 1, iw = owi * stride + kw - 1;
      if (ih < 0 || ih >= Hin || iw < 0 || iw >= Win) return 0.f;
      size_t idx = in_nchw ? (((size_t)biA * Cin + cin) * Hin + ih) * Win + iw
                           : (((size_t)biA * Hin + ih) * Win + iw) * Cin + cin;
      return in[idx];
    });
    v16h b = frag_b([&](int k, int n) -> float {
      int kk = k0 + k, nn = n0 + n;
      if (kk >= K || nn >= Cout) return 0.f;
      int pidx = kk / Cin, cin = kk - pidx * Cin;
      int kh = pidx / 3, kw = pidx - kh * 3;
      return w[(((size_t)nn * Cin + cin) * 3 + kh) * 3 + kw];
    });
    acc = wmma32(a, b, acc);
  }
  frag_c_store(acc, [&](int mr, int nl, float v) {
    int m = m0 + mr, n = n0 + nl;
    if (m >= M || n >= Cout) return;
    float sc = gam[n] * rsqrtf(var[n] + 1e-5f);
    v = (v + bias[n] - mu[n]) * sc + bet[n];
    outp[(size_t)m * Cout + n] = fmaxf(v, 0.f);
  });
}

// ---------------- generic GEMM: out = act(A[M,K] @ W[N,K]^T + bias) + res ----------------
// act: 0 none, 1 relu, 2 gelu(exact), 3 sigmoid

__global__ void k_gemm(const float* __restrict__ A, int lda,
                       const float* __restrict__ W, const float* __restrict__ bias,
                       const float* __restrict__ res, float* __restrict__ outp,
                       int ldo, int M, int N, int K, int act) {
  const int m0 = blockIdx.x * 16, n0 = blockIdx.y * 16;
  v8f acc = {};
  for (int k0 = 0; k0 < K; k0 += 32) {
    v16h a = frag_a([&](int m, int k) -> float {
      int mm = m0 + m, kk = k0 + k;
      return (mm < M && kk < K) ? A[(size_t)mm * lda + kk] : 0.f;
    });
    v16h b = frag_b([&](int k, int n) -> float {
      int nn = n0 + n, kk = k0 + k;
      return (nn < N && kk < K) ? W[(size_t)nn * K + kk] : 0.f;
    });
    acc = wmma32(a, b, acc);
  }
  frag_c_store(acc, [&](int mr, int nl, float v) {
    int m = m0 + mr, n = n0 + nl;
    if (m >= M || n >= N) return;
    if (bias) v += bias[n];
    if (act == 1) v = fmaxf(v, 0.f);
    else if (act == 2) v = 0.5f * v * (1.f + erff(v * 0.70710678118f));
    else if (act == 3) v = 1.f / (1.f + expf(-v));
    if (res) v += res[(size_t)m * ldo + n];
    outp[(size_t)m * ldo + n] = v;
  });
}

// ---------------- KV-projection GEMM with Tensor Data Mover A-strip staging ----------------
// K must be 256. Loads a 16x256 fp32 strip of A into LDS via TDM, then 8 WMMA K-steps.

__global__ void k_gemm_kv_tdm(const float* __restrict__ A, const float* __restrict__ W,
                              const float* __restrict__ bias, float* __restrict__ outp,
                              int M, int N, int K) {
  __shared__ float at[16 * 256];
  const int m0 = blockIdx.x * 16, n0 = blockIdx.y * 16;
#if __has_builtin(__builtin_amdgcn_tensor_load_to_lds) && __has_builtin(__builtin_amdgcn_s_wait_tensorcnt)
  {
    unsigned lds_off = (unsigned)(unsigned long long)(const void*)&at[0];
    unsigned long long ga = (unsigned long long)(const void*)(A + (size_t)m0 * K);
    unsigned td1 = (unsigned)(M - m0);     // remaining rows: OOB rows read as zero
    v4u g0;
    g0[0] = 1u;                                               // count = 1 (valid D#)
    g0[1] = lds_off;                                          // lds_addr (bytes)
    g0[2] = (unsigned)(ga & 0xffffffffu);                     // global_addr lo
    g0[3] = (unsigned)((ga >> 32) & 0x01ffffffu) | (2u << 30);// global_addr hi | type=2
    v8i_t g1;
    g1[0] = (int)(2u << 16);                                  // data_size = 4B
    g1[1] = (int)(((unsigned)K & 0xffffu) << 16);             // tensor_dim0[15:0]
    g1[2] = (int)((((unsigned)K >> 16) & 0xffffu) | ((td1 & 0xffffu) << 16));
    g1[3] = (int)(((td1 >> 16) & 0xffffu) | (((unsigned)K & 0xffffu) << 16)); // tile_dim0=K
    g1[4] = 16;                                               // tile_dim1 = 16 rows
    g1[5] = (int)(unsigned)K;                                 // tensor_dim0_stride lo32
    g1[6] = 0;
    g1[7] = 0;
    v4i_t gz = {0, 0, 0, 0};
    v8i_t gz8 = {0, 0, 0, 0, 0, 0, 0, 0};
    // 6-arg form on this toolchain: (g0, g1, g2, g3, extra group, cpol)
    __builtin_amdgcn_tensor_load_to_lds(g0, g1, gz, gz, gz8, 0);
    __builtin_amdgcn_s_wait_tensorcnt(0);
    asm volatile("" ::: "memory");
  }
#else
  for (int i = threadIdx.x; i < 16 * 256; i += 32) {
    int r = i >> 8, c = i & 255;
    int mm = m0 + r;
    at[i] = (mm < M) ? A[(size_t)mm * K + c] : 0.f;
  }
  __syncthreads();
#endif
  v8f acc = {};
  for (int k0 = 0; k0 < K; k0 += 32) {
    v16h a = frag_a([&](int m, int k) -> float { return at[m * 256 + k0 + k]; });
    v16h b = frag_b([&](int k, int n) -> float {
      int nn = n0 + n;
      return (nn < N) ? W[(size_t)nn * K + k0 + k] : 0.f;
    });
    acc = wmma32(a, b, acc);
  }
  frag_c_store(acc, [&](int mr, int nl, float v) {
    int m = m0 + mr, n = n0 + nl;
    if (m >= M || n >= N) return;
    outp[(size_t)m * N + n] = v + bias[n];
  });
}

// ---------------- attention: scores = scale * Q Kh^T (hd=32 -> one WMMA) ----------------

__global__ void k_scores(const float* __restrict__ Qm, int q_ld, int q_off,
                         const float* __restrict__ Km, int k_ld, int k_off,
                         float* __restrict__ scores, int Lq, int Lk, int NHp,
                         float scale) {
  const int b = blockIdx.z / NHp, h = blockIdx.z % NHp;
  const int q0 = blockIdx.x * 16, s0 = blockIdx.y * 16;
  v8f acc = {};
  v16h a = frag_a([&](int m, int k) -> float {
    int row = q0 + m;
    return (row < Lq) ? Qm[((size_t)b * Lq + row) * q_ld + q_off + h * 32 + k] : 0.f;
  });
  v16h bf = frag_b([&](int k, int n) -> float {
    int col = s0 + n;
    return (col < Lk) ? Km[((size_t)b * Lk + col) * k_ld + k_off + h * 32 + k] : 0.f;
  });
  acc = wmma32(a, bf, acc);
  frag_c_store(acc, [&](int mr, int nl, float v) {
    int row = q0 + mr, col = s0 + nl;
    if (row >= Lq || col >= Lk) return;
    scores[(((size_t)b * NHp + h) * Lq + row) * Lk + col] = v * scale;
  });
}

// ---------------- attention: O_head = P @ V_head ----------------

__global__ void k_av(const float* __restrict__ P, const float* __restrict__ Vm,
                     int v_ld, int v_off, float* __restrict__ O, int Lq, int Lk,
                     int NHp) {
  const int b = blockIdx.z / NHp, h = blockIdx.z % NHp;
  const int q0 = blockIdx.x * 16, n0 = blockIdx.y * 16;
  v8f acc = {};
  for (int k0 = 0; k0 < Lk; k0 += 32) {
    v16h a = frag_a([&](int m, int k) -> float {
      int row = q0 + m, kk = k0 + k;
      return (row < Lq && kk < Lk)
                 ? P[(((size_t)b * NHp + h) * Lq + row) * Lk + kk]
                 : 0.f;
    });
    v16h bf = frag_b([&](int k, int n) -> float {
      int kk = k0 + k;
      return (kk < Lk)
                 ? Vm[((size_t)b * Lk + kk) * v_ld + v_off + h * 32 + n0 + n]
                 : 0.f;
    });
    acc = wmma32(a, bf, acc);
  }
  frag_c_store(acc, [&](int mr, int nl, float v) {
    int row = q0 + mr;
    if (row >= Lq) return;
    O[((size_t)b * Lq + row) * 256 + h * 32 + n0 + nl] = v;
  });
}

// ---------------- row softmax (in place) ----------------

__global__ void k_softmax(float* __restrict__ x, int Slen) {
  const int t = threadIdx.x;
  float* p = x + (size_t)blockIdx.x * Slen;
  __shared__ float red[256];
  float mx = -3.4e38f;
  for (int i = t; i < Slen; i += 256) mx = fmaxf(mx, p[i]);
  red[t] = mx;
  __syncthreads();
  for (int s = 128; s > 0; s >>= 1) {
    if (t < s) red[t] = fmaxf(red[t], red[t + s]);
    __syncthreads();
  }
  mx = red[0];
  __syncthreads();
  float sum = 0.f;
  for (int i = t; i < Slen; i += 256) {
    float e = expf(p[i] - mx);
    p[i] = e;
    sum += e;
  }
  red[t] = sum;
  __syncthreads();
  for (int s = 128; s > 0; s >>= 1) {
    if (t < s) red[t] += red[t + s];
    __syncthreads();
  }
  float inv = 1.f / red[0];
  for (int i = t; i < Slen; i += 256) p[i] *= inv;
}

// ---------------- layernorm over D=256 ----------------

__global__ void k_ln(const float* __restrict__ x, const float* __restrict__ g,
                     const float* __restrict__ b, float* __restrict__ y) {
  const int t = threadIdx.x;  // 256
  const size_t base = (size_t)blockIdx.x * 256;
  __shared__ float red[256];
  float v = x[base + t];
  red[t] = v;
  __syncthreads();
  for (int s = 128; s > 0; s >>= 1) {
    if (t < s) red[t] += red[t + s];
    __syncthreads();
  }
  float mean = red[0] * (1.f / 256.f);
  __syncthreads();
  float d = v - mean;
  red[t] = d * d;
  __syncthreads();
  for (int s = 128; s > 0; s >>= 1) {
    if (t < s) red[t] += red[t + s];
    __syncthreads();
  }
  float rstd = rsqrtf(red[0] * (1.f / 256.f) + 1e-5f);
  y[base + t] = d * rstd * g[t] + b[t];
}

// ---------------- TTT: analytic grad of sum_b mean_q (1 - sigmoid(simhead(q))) ----------------

__global__ void k_ttt_grad(const float* __restrict__ x, const float* __restrict__ w1,
                           const float* __restrict__ b1, const float* __restrict__ w2,
                           const float* __restrict__ b2, float* __restrict__ grad) {
  const int t = threadIdx.x;  // 256
  const size_t row = blockIdx.x;
  __shared__ float qv[256];
  __shared__ float coef[256];
  __shared__ float red[256];
  __shared__ float cs;
  qv[t] = x[row * 256 + t];
  __syncthreads();
  float h = b1[t];
  const float* wr = w1 + (size_t)t * 256;
  for (int d = 0; d < 256; ++d) h += qv[d] * wr[d];
  float hr = fmaxf(h, 0.f);
  coef[t] = (h > 0.f) ? w2[t] : 0.f;
  red[t] = w2[t] * hr;
  __syncthreads();
  for (int s = 128; s > 0; s >>= 1) {
    if (t < s) red[t] += red[t + s];
    __syncthreads();
  }
  if (t == 0) {
    float logit = red[0] + b2[0];
    float sg = 1.f / (1.f + expf(-logit));
    cs = -sg * (1.f - sg) * 0.01f;  // d/dlogit of mean over Q=100 of (1 - sigmoid)
  }
  __syncthreads();
  float gsum = 0.f;
  for (int f = 0; f < 256; ++f) gsum += coef[f] * w1[(size_t)f * 256 + t];
  grad[row * 256 + t] = cs * gsum;
}

__global__ void k_ttt_update(float* __restrict__ vmt, const float* __restrict__ grad,
                             float* __restrict__ x, float* __restrict__ snap,
                             const float* __restrict__ rlr,
                             const float* __restrict__ rmom, int n) {
  int i = blockIdx.x * 256 + threadIdx.x;
  if (i >= n) return;
  float lr = log1pf(expf(rlr[0])) + 1e-8f;          // softplus
  float mom = 1.f / (1.f + expf(-rmom[0])) + 1e-8f; // sigmoid
  float v = mom * vmt[i] - lr * grad[i];
  vmt[i] = v;
  float nx = x[i] + v;
  x[i] = nx;
  snap[i] = nx;
}

// ---------------- small utilities ----------------

__global__ void k_copy(float* __restrict__ dst, const float* __restrict__ src, int n) {
  int i = blockIdx.x * 256 + threadIdx.x;
  if (i < n) dst[i] = src[i];
}
__global__ void k_zero(float* __restrict__ p, int n) {
  int i = blockIdx.x * 256 + threadIdx.x;
  if (i < n) p[i] = 0.f;
}
__global__ void k_addpos(float* __restrict__ mem, const float* __restrict__ pos, int n) {
  int i = blockIdx.x * 256 + threadIdx.x;
  if (i < n) {
    int d = i & 255;
    int s = (i >> 8) & 16383;
    mem[i] += pos[s * 256 + d];
  }
}
__global__ void k_broadcast_q(float* __restrict__ tgt, const float* __restrict__ qe, int n) {
  int i = blockIdx.x * 256 + threadIdx.x;
  if (i < n) {
    int d = i & 255;
    int q = (i >> 8) % 100;
    tgt[i] = qe[q * 256 + d];
  }
}
__global__ void k_signorm(float* __restrict__ x) {  // rows of 64
  const int t = threadIdx.x;
  float* p = x + (size_t)blockIdx.x * 64;
  __shared__ float red[64];
  float v = p[t];
  red[t] = v * v;
  __syncthreads();
  for (int s = 32; s > 0; s >>= 1) {
    if (t < s) red[t] += red[t + s];
    __syncthreads();
  }
  float nrm = sqrtf(red[0]);
  p[t] = v / fmaxf(nrm, 1e-12f);
}

// ---------------- host orchestration ----------------

extern "C" void kernel_launch(void* const* d_in, const int* in_sizes, int n_in,
                              void* d_out, int out_size, void* d_ws, size_t ws_size,
                              hipStream_t stream) {
  (void)in_sizes; (void)n_in; (void)out_size; (void)ws_size;
  auto F = [&](int i) { return (const float*)d_in[i]; };
  const float* images = F(0);
  const float* pos = F(19);
  const float* qe = F(20);

  float* ws = (float*)d_ws;
  float* out = (float*)d_out;

  const int B_ = 8, D_ = 256, S_ = 16384, Qn = 100, NHn = 8, DFFn = 1024;
  const int TOK = B_ * Qn * D_;  // 204800

  size_t off = 0;
  float* mem = ws + off; off += (size_t)B_ * S_ * D_;        // 33.5M f
  float* kv  = ws + off; off += (size_t)B_ * S_ * 2 * D_;    // 67.1M f
  float* sc  = ws + off; off += (size_t)B_ * NHn * Qn * S_;  // 104.9M f
  float* x1 = sc;                 // conv1 out (overlaid, used pre-attention only)
  float* x2 = sc + 16777216;      // conv2 out (overlaid)
  float* tgt  = ws + off; off += TOK;
  float* tbuf = ws + off; off += TOK;
  float* qkv  = ws + off; off += (size_t)TOK * 3;
  float* qb   = ws + off; off += TOK;
  float* obuf = ws + off; off += TOK;
  float* ffn  = ws + off; off += (size_t)B_ * Qn * DFFn;
  float* vmt  = ws + off; off += TOK;
  float* grad = ws + off; off += TOK;
  float* hbuf = ws + off; off += TOK;

  float* o_mask = out;
  float* o_cls  = out + 204800;
  float* o_sig  = out + 284800;
  float* o_sim  = out + 336000;
  float* o_mar  = out + 336800;
  float* o_ttt  = out + 337600;

  auto gemm = [&](const float* A, int lda, const float* W, const float* bias,
                  const float* res, float* O, int ldo, int M, int N, int K, int act) {
    dim3 g((M + 15) / 16, (N + 15) / 16, 1);
    k_gemm<<<g, 32, 0, stream>>>(A, lda, W, bias, res, O, ldo, M, N, K, act);
  };

  // ---- backbone (implicit GEMM + fused BN/ReLU, NHWC) ----
  k_conv<<<dim3(524288 / 16, 2), 32, 0, stream>>>(images, 512, 512, 3, 1, F(1), F(2),
                                                  F(3), F(4), F(5), F(6), x1, 256, 256,
                                                  32, 2, B_);
  k_conv<<<dim3(131072 / 16, 4), 32, 0, stream>>>(x1, 256, 256, 32, 0, F(7), F(8), F(9),
                                                  F(10), F(11), F(12), x2, 128, 128, 64,
                                                  2, B_);
  k_conv<<<dim3(131072 / 16, 16), 32, 0, stream>>>(x2, 128, 128, 64, 0, F(13), F(14),
                                                   F(15), F(16), F(17), F(18), mem, 128,
                                                   128, 256, 1, B_);
  {
    int n = B_ * S_ * D_;
    k_addpos<<<(n + 255) / 256, 256, 0, stream>>>(mem, pos, n);
  }
  k_broadcast_q<<<(TOK + 255) / 256, 256, 0, stream>>>(tgt, qe, TOK);

  // detached sim-head params used by the TTT inner loop
  const float* sw1 = F(153); const float* sb1 = F(154);
  const float* sw2 = F(155); const float* sb2 = F(156);

  for (int l = 0; l < 6; ++l) {
    const int L0 = 21 + 20 * l;
    // --- self-attention ---
    k_ln<<<800, 256, 0, stream>>>(tgt, F(L0 + 0), F(L0 + 1), tbuf);
    gemm(tbuf, 256, F(L0 + 6), F(L0 + 7), nullptr, qkv, 768, 800, 768, 256, 0);
    k_scores<<<dim3(7, 7, 64), 32, 0, stream>>>(qkv, 768, 0, qkv, 768, 256, sc, Qn, Qn,
                                                NHn, 0.176776695f);
    k_softmax<<<6400, 256, 0, stream>>>(sc, Qn);
    k_av<<<dim3(7, 2, 64), 32, 0, stream>>>(sc, qkv, 768, 512, obuf, Qn, Qn, NHn);
    gemm(obuf, 256, F(L0 + 8), F(L0 + 9), tgt, tgt, 256, 800, 256, 256, 0);
    // --- cross-attention ---
    k_ln<<<800, 256, 0, stream>>>(tgt, F(L0 + 2), F(L0 + 3), tbuf);
    gemm(tbuf, 256, F(L0 + 10), F(L0 + 11), nullptr, qb, 256, 800, 256, 256, 0);
    k_gemm_kv_tdm<<<dim3(8192, 32), 32, 0, stream>>>(mem, F(L0 + 10) + 65536,
                                                     F(L0 + 11) + 256, kv, 131072, 512,
                                                     256);
    k_scores<<<dim3(7, 1024, 64), 32, 0, stream>>>(qb, 256, 0, kv, 512, 0, sc, Qn, S_,
                                                   NHn, 0.176776695f);
    k_softmax<<<6400, 256, 0, stream>>>(sc, S_);
    k_av<<<dim3(7, 2, 64), 32, 0, stream>>>(sc, kv, 512, 256, obuf, Qn, S_, NHn);
    gemm(obuf, 256, F(L0 + 12), F(L0 + 13), tgt, tgt, 256, 800, 256, 256, 0);
    // --- TTT inner GD-with-momentum ---
    k_copy<<<(TOK + 255) / 256, 256, 0, stream>>>(o_ttt + (size_t)(l * 4) * TOK, tgt, TOK);
    k_zero<<<(TOK + 255) / 256, 256, 0, stream>>>(vmt, TOK);
    for (int st = 0; st < 3; ++st) {
      k_ttt_grad<<<800, 256, 0, stream>>>(tgt, sw1, sb1, sw2, sb2, grad);
      float* snap = o_ttt + (size_t)(l * 4 + st + 1) * TOK;
      k_ttt_update<<<(TOK + 255) / 256, 256, 0, stream>>>(vmt, grad, tgt, snap,
                                                          F(L0 + 18), F(L0 + 19), TOK);
    }
    // --- FFN ---
    k_ln<<<800, 256, 0, stream>>>(tgt, F(L0 + 4), F(L0 + 5), tbuf);
    gemm(tbuf, 256, F(L0 + 14), F(L0 + 15), nullptr, ffn, 1024, 800, 1024, 256, 2);
    gemm(ffn, 1024, F(L0 + 16), F(L0 + 17), tgt, tgt, 256, 800, 256, 1024, 0);
  }

  // ---- heads ----
  gemm(tgt, 256, F(141), F(142), nullptr, hbuf, 256, 800, 256, 256, 1);
  gemm(hbuf, 256, F(143), F(144), nullptr, o_mask, 256, 800, 256, 256, 0);
  gemm(tgt, 256, F(145), F(146), nullptr, hbuf, 256, 800, 256, 256, 1);
  gemm(hbuf, 256, F(147), F(148), nullptr, o_cls, 100, 800, 100, 256, 0);
  gemm(tgt, 256, F(149), F(150), nullptr, hbuf, 256, 800, 256, 256, 1);
  gemm(hbuf, 256, F(151), F(152), nullptr, o_sig, 64, 800, 64, 256, 0);
  k_signorm<<<800, 64, 0, stream>>>(o_sig);
  gemm(tgt, 256, F(153), F(154), nullptr, hbuf, 256, 800, 256, 256, 1);
  gemm(hbuf, 256, F(155), F(156), nullptr, o_sim, 1, 800, 1, 256, 3);
  gemm(tgt, 256, F(157), F(158), nullptr, hbuf, 256, 800, 256, 256, 1);
  gemm(hbuf, 256, F(159), F(160), nullptr, o_mar, 1, 800, 1, 256, 3);
}
